// FullAttention_21406117004198
// MI455X (gfx1250) — matrix operand order repaired and checked by reading, not compiled, not run on hardware
//
#include <hip/hip_runtime.h>
#include <hip/hip_bf16.h>

typedef __bf16 v16bf __attribute__((ext_vector_type(16)));
typedef float  v8f   __attribute__((ext_vector_type(8)));

#define NEG_INF (-__builtin_inff())

constexpr int Bb = 8;     // batch
constexpr int Ll = 2048;  // query length
constexpr int Ss = 2048;  // key length
constexpr int HE = 512;   // H*E merged contraction dim == H*D output width
constexpr int BM = 64;    // query rows per block
constexpr int BN = 64;    // key rows per chunk
constexpr int KSTR = 520; // Klds row stride (halves): 1040B -> bank step 4, conflict-free b128
constexpr int VSTR = 72;  // Vlds col stride (halves): 144B  -> bank step 36, conflict-free b128
constexpr int PSTR = 72;  // Plds row stride (halves)
constexpr float SCALE = 0.125f; // 1/sqrt(E=64)

// LDS layout (bytes)
constexpr int OFF_K = 0;                         // 64*520*2 = 66560
constexpr int OFF_V = 66560;                     // 512*72*2 = 73728
constexpr int OFF_P = 66560 + 73728;             // 140288 ; 64*72*2 = 9216
constexpr int OFF_ST = 140288 + 9216;            // 149504 ; 448 floats
constexpr int SMEM_BYTES = 149504 + 448 * 4;     // 151296

union AFrag {
    v16bf v;
    __bf16 h[16];
    uint4 u[2];
};

__device__ inline void cvt4(__bf16* h, float4 f) {
    h[0] = (__bf16)f.x; h[1] = (__bf16)f.y; h[2] = (__bf16)f.z; h[3] = (__bf16)f.w;
}

__global__ __launch_bounds__(256)
void fattn_fwd_kernel(const float* __restrict__ Q, const float* __restrict__ K,
                      const float* __restrict__ V, const unsigned char* __restrict__ Msk,
                      float* __restrict__ O)
{
    extern __shared__ __align__(16) char smem[];
    __bf16* Klds = (__bf16*)(smem + OFF_K);     // [BN][KSTR] row-major bf16
    __bf16* Vlds = (__bf16*)(smem + OFF_V);     // [HE][VSTR] col-major bf16 (V^T)
    __bf16* Plds = (__bf16*)(smem + OFF_P);     // [BM][PSTR] row-major bf16
    float*  st   = (float*)(smem + OFF_ST);
    float* m_arr = st;        // [64] running row max
    float* l_arr = st + 64;   // [64] running row sum
    float* a_arr = st + 128;  // [64] per-chunk rescale alpha
    float* cmax2 = st + 192;  // [2][64] per-colgroup row max
    float* csum2 = st + 320;  // [2][64] per-colgroup row sum

    const int tid  = threadIdx.x;
    const int wave = tid >> 5;
    const int lane = tid & 31;
    const int lo   = lane & 15;
    const int hi   = lane >> 4;   // 0 or 1
    const int r    = wave & 3;    // 16-row tile within block
    const int g    = wave >> 2;   // output column half (0:cols 0-255, 1:cols 256-511)

    const int b  = blockIdx.y;
    const int i0 = blockIdx.x * BM;

    // ---- Q block -> per-wave bf16 A-fragments in registers (K = 0..511, 16 frags) ----
    // 16-bit A layout: lane(lo)=row; lanes<16 hold K {k0..k0+7, k0+16..k0+23} with k0=kb*32;
    // lanes>=16 same rows with k0=kb*32+8.
    AFrag qf[16];
    {
        const float* qp = Q + ((size_t)b * Ll + i0 + r * 16 + lo) * HE;
        #pragma unroll
        for (int kb = 0; kb < 16; ++kb) {
            const int k0 = kb * 32 + hi * 8;
            float4 f0 = *(const float4*)(qp + k0);
            float4 f1 = *(const float4*)(qp + k0 + 4);
            float4 f2 = *(const float4*)(qp + k0 + 16);
            float4 f3 = *(const float4*)(qp + k0 + 20);
            cvt4(&qf[kb].h[0],  f0);
            cvt4(&qf[kb].h[4],  f1);
            cvt4(&qf[kb].h[8],  f2);
            cvt4(&qf[kb].h[12], f3);
        }
    }

    // ---- output accumulators: 16 col-tiles x (16x16 f32 C fragment) ----
    v8f acc[16];
    #pragma unroll
    for (int t = 0; t < 16; ++t)
        #pragma unroll
        for (int jj = 0; jj < 8; ++jj) acc[t][jj] = 0.f;

    if (tid < 64) { m_arr[tid] = NEG_INF; l_arr[tid] = 0.f; }

    for (int j0 = 0; j0 < Ss; j0 += BN) {
        __syncthreads(); // B0: previous chunk fully consumed; LDS reusable

        // ---- stage K chunk (row-major bf16) ----
        #pragma unroll 4
        for (int it = 0; it < 32; ++it) {
            const int idx = it * 256 + tid;     // 64 rows * 128 float4-groups
            const int row = idx >> 7;
            const int c4  = (idx & 127) << 2;
            float4 f = *(const float4*)(K + ((size_t)b * Ss + j0 + row) * HE + c4);
            union { __bf16 h[4]; uint2 u2; } pk;
            cvt4(pk.h, f);
            *(uint2*)(Klds + row * KSTR + c4) = pk.u2;
        }
        // ---- stage V chunk transposed (col-major bf16) ----
        #pragma unroll 4
        for (int it = 0; it < 32; ++it) {
            const int idx = it * 256 + tid;
            const int s  = idx >> 7;
            const int d4 = (idx & 127) << 2;
            float4 f = *(const float4*)(V + ((size_t)b * Ss + j0 + s) * HE + d4);
            Vlds[(d4 + 0) * VSTR + s] = (__bf16)f.x;
            Vlds[(d4 + 1) * VSTR + s] = (__bf16)f.y;
            Vlds[(d4 + 2) * VSTR + s] = (__bf16)f.z;
            Vlds[(d4 + 3) * VSTR + s] = (__bf16)f.w;
        }
        if (tid < 128) { cmax2[tid] = NEG_INF; csum2[tid] = 0.f; }
        __syncthreads(); // B1: K/V staged

        // ---- scores: two 16x16 tiles per wave, cols g*32 .. g*32+31 of chunk ----
        v8f sc0, sc1;
        #pragma unroll
        for (int jj = 0; jj < 8; ++jj) { sc0[jj] = 0.f; sc1[jj] = 0.f; }
        #pragma unroll
        for (int kb = 0; kb < 16; ++kb) {
            const int kk = kb * 32 + hi * 8;    // B frag: K {kk..kk+7, kk+16..kk+23} per lane
            AFrag bf0, bf1;
            const __bf16* kp0 = Klds + (g * 32 + lo) * KSTR + kk;
            const __bf16* kp1 = Klds + (g * 32 + 16 + lo) * KSTR + kk;
            bf0.u[0] = *(const uint4*)kp0; bf0.u[1] = *(const uint4*)(kp0 + 16);
            bf1.u[0] = *(const uint4*)kp1; bf1.u[1] = *(const uint4*)(kp1 + 16);
            sc0 = __builtin_amdgcn_wmma_f32_16x16x32_bf16(false, qf[kb].v, false, bf0.v,
                                                          (short)0, sc0, false, false);
            sc1 = __builtin_amdgcn_wmma_f32_16x16x32_bf16(false, qf[kb].v, false, bf1.v,
                                                          (short)0, sc1, false, false);
        }

        // ---- scale, mask, per-row max (rows live across 16 lanes of one VGPR) ----
        const unsigned char* mbase =
            Msk + ((size_t)b * Ll + i0 + r * 16) * (size_t)Ss + j0 + g * 32 + lo;
        float rmx[8];
        #pragma unroll
        for (int jj = 0; jj < 8; ++jj) {
            const int mr = jj + 8 * hi;
            const unsigned char* mp = mbase + (size_t)mr * Ss;
            float s0 = sc0[jj] * SCALE;
            float s1 = sc1[jj] * SCALE;
            if (mp[0])  s0 = NEG_INF;
            if (mp[16]) s1 = NEG_INF;
            sc0[jj] = s0; sc1[jj] = s1;
            float v = fmaxf(s0, s1);
            v = fmaxf(v, __shfl_xor(v, 1, 16));
            v = fmaxf(v, __shfl_xor(v, 2, 16));
            v = fmaxf(v, __shfl_xor(v, 4, 16));
            v = fmaxf(v, __shfl_xor(v, 8, 16));
            rmx[jj] = v;
        }
        if (lo == 0) {
            #pragma unroll
            for (int jj = 0; jj < 8; ++jj)
                cmax2[g * 64 + r * 16 + jj + 8 * hi] = rmx[jj];
        }
        __syncthreads(); // B2: chunk maxima visible

        if (wave == 0) {
            #pragma unroll
            for (int q2 = 0; q2 < 2; ++q2) {
                const int rr = lane + q2 * 32;
                float cm = fmaxf(cmax2[rr], cmax2[64 + rr]);
                float mp = m_arr[rr];
                float mn = fmaxf(mp, cm);
                float al = (mn > NEG_INF) ? __expf(mp - mn) : 0.f;
                m_arr[rr] = mn;
                a_arr[rr] = al;
            }
        }
        __syncthreads(); // B3: m/alpha ready

        // ---- P = exp(s - m), row sums, stage P to LDS as bf16 ----
        float psum[8];
        #pragma unroll
        for (int jj = 0; jj < 8; ++jj) {
            const int row = r * 16 + jj + 8 * hi;
            const float mn = m_arr[row];
            float s0 = sc0[jj], s1 = sc1[jj];
            float p0 = (s0 > NEG_INF) ? __expf(s0 - mn) : 0.f;
            float p1 = (s1 > NEG_INF) ? __expf(s1 - mn) : 0.f;
            Plds[row * PSTR + g * 32 + lo]      = (__bf16)p0;
            Plds[row * PSTR + g * 32 + 16 + lo] = (__bf16)p1;
            float v = p0 + p1;
            v += __shfl_xor(v, 1, 16);
            v += __shfl_xor(v, 2, 16);
            v += __shfl_xor(v, 4, 16);
            v += __shfl_xor(v, 8, 16);
            psum[jj] = v;
        }
        if (lo == 0) {
            #pragma unroll
            for (int jj = 0; jj < 8; ++jj)
                csum2[g * 64 + r * 16 + jj + 8 * hi] = psum[jj];
        }
        __syncthreads(); // B4: P + sums staged

        if (wave == 0) {
            #pragma unroll
            for (int q2 = 0; q2 < 2; ++q2) {
                const int rr = lane + q2 * 32;
                l_arr[rr] = l_arr[rr] * a_arr[rr] + csum2[rr] + csum2[64 + rr];
            }
        }

        // ---- rescale accumulators by alpha ----
        float al[8];
        #pragma unroll
        for (int jj = 0; jj < 8; ++jj)
            al[jj] = a_arr[r * 16 + jj + 8 * hi];
        #pragma unroll
        for (int t = 0; t < 16; ++t)
            #pragma unroll
            for (int jj = 0; jj < 8; ++jj)
                acc[t][jj] *= al[jj];

        // ---- O += P @ V_chunk ----
        AFrag pf[2];
        #pragma unroll
        for (int kb2 = 0; kb2 < 2; ++kb2) {
            const __bf16* pp = Plds + (r * 16 + lo) * PSTR + kb2 * 32 + hi * 8;
            pf[kb2].u[0] = *(const uint4*)pp;
            pf[kb2].u[1] = *(const uint4*)(pp + 16);
        }
        #pragma unroll
        for (int t = 0; t < 16; ++t) {
            #pragma unroll
            for (int kb2 = 0; kb2 < 2; ++kb2) {
                AFrag vf;
                const __bf16* vp = Vlds + (g * 256 + t * 16 + lo) * VSTR + kb2 * 32 + hi * 8;
                vf.u[0] = *(const uint4*)vp;
                vf.u[1] = *(const uint4*)(vp + 16);
                acc[t] = __builtin_amdgcn_wmma_f32_16x16x32_bf16(false, pf[kb2].v, false, vf.v,
                                                                 (short)0, acc[t], false, false);
            }
        }
    }

    __syncthreads(); // final l_arr ready

    // ---- normalize and store ----
    float rinv[8];
    #pragma unroll
    for (int jj = 0; jj < 8; ++jj) {
        float lv = l_arr[r * 16 + jj + 8 * hi];
        rinv[jj] = (lv > 0.f) ? (1.f / lv) : 0.f;
    }
    float* op = O + ((size_t)b * Ll + i0 + r * 16) * HE + g * 256;
    #pragma unroll
    for (int t = 0; t < 16; ++t) {
        #pragma unroll
        for (int jj = 0; jj < 8; ++jj) {
            op[(size_t)(jj + 8 * hi) * HE + t * 16 + lo] = acc[t][jj] * rinv[jj];
        }
    }
}

extern "C" void kernel_launch(void* const* d_in, const int* in_sizes, int n_in,
                              void* d_out, int out_size, void* d_ws, size_t ws_size,
                              hipStream_t stream) {
    (void)in_sizes; (void)n_in; (void)out_size; (void)d_ws; (void)ws_size;
    const float* Q = (const float*)d_in[0];
    const float* K = (const float*)d_in[1];
    const float* V = (const float*)d_in[2];
    const unsigned char* M = (const unsigned char*)d_in[3];
    float* O = (float*)d_out;

    (void)hipFuncSetAttribute((const void*)fattn_fwd_kernel,
                              hipFuncAttributeMaxDynamicSharedMemorySize, SMEM_BYTES);

    dim3 grid(Ll / BM, Bb);
    fattn_fwd_kernel<<<grid, dim3(256), SMEM_BYTES, stream>>>(Q, K, V, M, O);
}
